// TemporalInteraction_41704132444271
// MI455X (gfx1250) — matrix operand order repaired and checked by reading, not compiled, not run on hardware
//
#include <hip/hip_runtime.h>
#include <stdint.h>

typedef __attribute__((ext_vector_type(16))) _Float16 v16h;
typedef __attribute__((ext_vector_type(8)))  float    v8f;
typedef __attribute__((ext_vector_type(4)))  int      v4i;

static __device__ __forceinline__ unsigned pk2h(float a, float b) {
  _Float16 ha = (_Float16)a, hb = (_Float16)b;
  return (unsigned)__builtin_bit_cast(unsigned short, ha) |
         ((unsigned)__builtin_bit_cast(unsigned short, hb) << 16);
}

// ---------------------------------------------------------------------------
// Pack a row-major f32 [512,512] weight into f16 WMMA B-fragments.
// Fragment (kt,nt): K=kt*32..+31, N=nt*16..+15.  ISA 16-bit B layout:
// lane half h (N = lane&15): dwords 0-3 hold K-offset 8h..8h+7, dwords 4-7 hold
// K-offset 16+8h..16+8h+7 (f16 pairs).  Each lane's 32 B stored
// contiguously: dword index = frag*256 + lane*8 + v.
// ---------------------------------------------------------------------------
__global__ __launch_bounds__(256) void ti_pack(const float* __restrict__ W,
                                               unsigned* __restrict__ dst) {
  const int frag = blockIdx.x;            // kt*32 + nt
  const int kt   = frag >> 5;
  const int nt   = frag & 31;
  const int tid  = threadIdx.x;
  const int lane = tid >> 3;
  const int v    = tid & 7;
  const int k = kt * 32 + ((lane >> 4) * 8) + ((v >> 2) * 16) + 2 * (v & 3);
  const int n = nt * 16 + (lane & 15);
  dst[(size_t)frag * 256 + tid] =
      pk2h(W[(size_t)k * 512 + n], W[(size_t)(k + 1) * 512 + n]);
}

__global__ __launch_bounds__(256) void ti_zero(float* __restrict__ p) {
  p[(size_t)blockIdx.x * 256 + threadIdx.x] = 0.0f;
}

// ---------------------------------------------------------------------------
// Y[M,512] = X[M,512] @ Wpack + bias  (X f32 -> f16 on the fly).
// Grid (M/32, 2); 8 waves; wave = 16x64 tile (4 accumulator subtiles).
// ---------------------------------------------------------------------------
__global__ __launch_bounds__(256) void ti_gemm512(const float* __restrict__ X,
                                                  const unsigned* __restrict__ Wp,
                                                  const float* __restrict__ bias,
                                                  float* __restrict__ Y) {
  const int tid  = threadIdx.x;
  const int lane = tid & 31;
  const int w    = tid >> 5;
  const int rowbase = blockIdx.x * 32 + (w & 1) * 16;
  const int colbase = blockIdx.y * 256 + (w >> 1) * 64;
  const int rl = lane & 15;
  const int hs = lane >> 4;

  v8f zero = {};
  v8f acc[4] = {zero, zero, zero, zero};

  const float* arow = X + (size_t)(rowbase + rl) * 512 + hs * 8;

  for (int ks = 0; ks < 16; ++ks) {
    const float* ap = arow + ks * 32;
    v16h a;
#pragma unroll
    for (int e = 0; e < 8; ++e) {
      a[e]     = (_Float16)ap[e];
      a[e + 8] = (_Float16)ap[e + 16];
    }
#pragma unroll
    for (int t = 0; t < 4; ++t) {
      const unsigned* bpk = Wp + ((size_t)(ks * 32 + (colbase >> 4) + t) * 256 + lane * 8);
      v16h bf;
      ((v4i*)&bf)[0] = *(const v4i*)bpk;
      ((v4i*)&bf)[1] = *(const v4i*)(bpk + 4);
      acc[t] = __builtin_amdgcn_wmma_f32_16x16x32_f16(false, a, false, bf,
                                                      (short)0, acc[t], false, false);
    }
  }

#pragma unroll
  for (int t = 0; t < 4; ++t) {
    const int col = colbase + t * 16 + rl;
    const float bv = bias[col];
    float* yp = Y + (size_t)(rowbase + hs * 8) * 512 + col;
#pragma unroll
    for (int r = 0; r < 8; ++r) yp[(size_t)r * 512] = acc[t][r] + bv;
  }
}

// ---------------------------------------------------------------------------
// Fused per-(b,i) kernel, single pass over all 64 j-rows.
//  Stage A: Abuf = relu(rel2@W_t + b_t + xe[b,j,:]) stored as f16 in
//           fragment-packed layout (frag(mf,ks) = 1 KB; lane chunk = 32 B,
//           16 B halves xor-swizzled by lane bit 3 -> conflict-free b128 reads)
//  Stage B: each wave owns a 64(M) x 64(N) tile: acc[16] v8f; per K-step
//           4 LDS A-frags + 4 L2 B-frags -> 16 WMMAs.  W_imp is read exactly
//           once per block (512 KB, L2-resident).
//  Epilogue: sigmoid * mask * values fused; full-j logsumexp in registers +
//           one shfl_xor(16); row-sums for mean(axis=3) via xor-reduce +
//           global_atomic_add_f32 into pre-zeroed out2.
// ---------------------------------------------------------------------------
__global__ __launch_bounds__(256, 1) void ti_fused(
    const float* __restrict__ x, const float* __restrict__ ts,
    const float* __restrict__ amask, const float* __restrict__ Wt,
    const float* __restrict__ bt, const float* __restrict__ bimp,
    const unsigned* __restrict__ Wp, const float* __restrict__ values,
    const float* __restrict__ xe, float* __restrict__ out1,
    float* __restrict__ out2) {
  __shared__ v4i Abuf[4096];   // exactly 64 KB

  const int blk = blockIdx.x;
  const int b   = blk >> 6;
  const int i   = blk & 63;
  const int b64 = b * 64;
  const int tid  = threadIdx.x;
  const int lane = tid & 31;
  const int rl = lane & 15;
  const int hs = lane >> 4;

  const float ti_v = ts[b64 + i];

  // ---- stage A: thread -> row j = tid&63, c-range = (tid>>6)*128 .. +127 ----
  {
    const int j  = tid & 63;
    const int c0 = (tid >> 6) * 128;
    const float tj = ts[b64 + j];
    const float r  = ti_v - tj;
    const float a0 = log1pf(fmaxf(r, 0.0f));
    const float a1 = log1pf(fmaxf(-r, 0.0f));
    const float* xep = xe + (size_t)(b64 + j) * 512;
    const int mf   = j >> 4;
    const int lb   = j & 15;
    const int swzj = (j >> 3) & 1;
#pragma unroll 4
    for (int q = 0; q < 16; ++q) {
      const int c = c0 + q * 8;
      float h[8];
#pragma unroll
      for (int e = 0; e < 8; ++e) {
        float t = fmaf(a0, Wt[c + e], fmaf(a1, Wt[512 + c + e], bt[c + e] + xep[c + e]));
        h[e] = fmaxf(t, 0.0f);
      }
      v4i pk;
      pk[0] = pk2h(h[0], h[1]);
      pk[1] = pk2h(h[2], h[3]);
      pk[2] = pk2h(h[4], h[5]);
      pk[3] = pk2h(h[6], h[7]);
      const int ks    = c >> 5;
      const int sub   = c & 31;
      const int hsw   = (sub >> 3) & 1;   // which half-wave lane group holds it
      const int phase = (sub >> 4) & 1;   // which 16B chunk of the lane
      const int lanew = lb | (hsw << 4);
      Abuf[(mf * 16 + ks) * 64 + lanew * 2 + (phase ^ swzj)] = pk;
    }
  }
  __syncthreads();

  // ---- stage B: wave w owns cols w*64 .. w*64+63, all 64 rows ----
  const int w   = tid >> 5;
  const int nc0 = w * 64;
  const int swz = (lane >> 3) & 1;

  v8f zero = {};
  v8f acc[16];
#pragma unroll
  for (int t = 0; t < 16; ++t) acc[t] = zero;

  for (int ks = 0; ks < 16; ++ks) {
    v16h a[4];
#pragma unroll
    for (int mf = 0; mf < 4; ++mf) {
      const v4i* fp = Abuf + (mf * 16 + ks) * 64 + lane * 2;
      ((v4i*)&a[mf])[0] = fp[swz];
      ((v4i*)&a[mf])[1] = fp[swz ^ 1];
    }
#pragma unroll
    for (int nt = 0; nt < 4; ++nt) {
      const unsigned* bpk = Wp + ((size_t)(ks * 32 + (nc0 >> 4) + nt) * 256 + lane * 8);
      v16h bf;
      ((v4i*)&bf)[0] = *(const v4i*)bpk;
      ((v4i*)&bf)[1] = *(const v4i*)(bpk + 4);
#pragma unroll
      for (int mf = 0; mf < 4; ++mf)
        acc[nt * 4 + mf] = __builtin_amdgcn_wmma_f32_16x16x32_f16(
            false, a[mf], false, bf, (short)0, acc[nt * 4 + mf], false, false);
    }
  }

  // ---- fused epilogue ----
  float bn[4];
#pragma unroll
  for (int nt = 0; nt < 4; ++nt) bn[nt] = bimp[nc0 + nt * 16 + rl];

  float lm[4], ls[4];
#pragma unroll
  for (int nt = 0; nt < 4; ++nt) { lm[nt] = -3.0e38f; ls[nt] = 0.0f; }

  const float* vbase = values + (size_t)b64 * 512;
  const float inv512 = 1.0f / 512.0f;

#pragma unroll
  for (int mf = 0; mf < 4; ++mf) {
    float maskr[8], rs[8];
#pragma unroll
    for (int r = 0; r < 8; ++r) {
      maskr[r] = amask[b64 + mf * 16 + r + hs * 8];
      rs[r] = 0.0f;
    }
#pragma unroll
    for (int nt = 0; nt < 4; ++nt) {
      const int n = nc0 + nt * 16 + rl;
      float p[8];
#pragma unroll
      for (int r = 0; r < 8; ++r) {
        float g = acc[nt * 4 + mf][r] + bn[nt];
        float s = 1.0f / (1.0f + __expf(-g));
        s *= maskr[r];
        rs[r] += s;
        p[r] = s * vbase[(size_t)(mf * 16 + r + hs * 8) * 512 + n];
      }
      float mx = p[0];
#pragma unroll
      for (int r = 1; r < 8; ++r) mx = fmaxf(mx, p[r]);
      float se = 0.0f;
#pragma unroll
      for (int r = 0; r < 8; ++r) se += __expf(p[r] - mx);
      float M = fmaxf(lm[nt], mx);
      ls[nt] = ls[nt] * __expf(lm[nt] - M) + se * __expf(mx - M);
      lm[nt] = M;
    }
    // row-sum (importances.mean over n): reduce across 16 lanes of each half
#pragma unroll
    for (int r = 0; r < 8; ++r) {
      float v = rs[r];
      v += __shfl_xor(v, 1, 32);
      v += __shfl_xor(v, 2, 32);
      v += __shfl_xor(v, 4, 32);
      v += __shfl_xor(v, 8, 32);
      if (rl == r)
        atomicAdd(out2 + (size_t)(b64 + i) * 64 + (mf * 16 + r + hs * 8), v * inv512);
    }
  }

  // combine the two half-wave row groups and write out1
  const float* xrow = x + (size_t)(b64 + i) * 512;
#pragma unroll
  for (int nt = 0; nt < 4; ++nt) {
    float mo = __shfl_xor(lm[nt], 16, 32);
    float so = __shfl_xor(ls[nt], 16, 32);
    float M = fmaxf(lm[nt], mo);
    float S = ls[nt] * __expf(lm[nt] - M) + so * __expf(mo - M);
    if (lane < 16) {
      const int n = nc0 + nt * 16 + rl;
      out1[(size_t)(b64 + i) * 512 + n] = xrow[n] + M + __logf(S);
    }
  }
}

// ---------------------------------------------------------------------------
extern "C" void kernel_launch(void* const* d_in, const int* in_sizes, int n_in,
                              void* d_out, int out_size, void* d_ws, size_t ws_size,
                              hipStream_t stream) {
  (void)in_sizes; (void)n_in; (void)out_size; (void)ws_size;
  const float* x     = (const float*)d_in[0];
  const float* ts    = (const float*)d_in[1];
  const float* amask = (const float*)d_in[2];
  const float* Wproj = (const float*)d_in[3];
  const float* bproj = (const float*)d_in[4];
  const float* Wx    = (const float*)d_in[5];
  const float* bx    = (const float*)d_in[6];
  const float* Wt    = (const float*)d_in[7];
  const float* bt    = (const float*)d_in[8];
  const float* Wimp  = (const float*)d_in[9];
  const float* bimp  = (const float*)d_in[10];

  float* out1 = (float*)d_out;                      // [16,64,512]
  float* out2 = out1 + (size_t)16 * 64 * 512;       // [16,64,64]

  unsigned* packProj = (unsigned*)d_ws;             // 512 KB each
  unsigned* packX    = packProj + 131072;
  unsigned* packImp  = packX + 131072;
  float* values      = (float*)(packImp + 131072);  // [1024,512] f32
  float* xe          = values + (size_t)1024 * 512; // [1024,512] f32

  ti_zero<<<dim3(256), dim3(256), 0, stream>>>(out2);          // 65536 floats
  ti_pack<<<dim3(512), dim3(256), 0, stream>>>(Wproj, packProj);
  ti_pack<<<dim3(512), dim3(256), 0, stream>>>(Wx, packX);
  ti_pack<<<dim3(512), dim3(256), 0, stream>>>(Wimp, packImp);

  ti_gemm512<<<dim3(32, 2), dim3(256), 0, stream>>>(x, packProj, bproj, values);
  ti_gemm512<<<dim3(32, 2), dim3(256), 0, stream>>>(x, packX, bx, xe);

  ti_fused<<<dim3(1024), dim3(256), 0, stream>>>(x, ts, amask, Wt, bt, bimp,
                                                 packImp, values, xe, out1, out2);
}
